// WavGCRN_58815282152225
// MI455X (gfx1250) — matrix operand order repaired and checked by reading, compiled, not run on hardware
//
#include <hip/hip_runtime.h>

typedef __attribute__((ext_vector_type(16))) __bf16 v16bf;
typedef __attribute__((ext_vector_type(8)))  __bf16 v8bf;
typedef __attribute__((ext_vector_type(8)))  float  v8f;

#define BATCH   64
#define NPTS    512
#define HIDC    64
#define SEQLEN  12
#define HALF    6
#define OUTLEN  12
#define DINP    96          // 68 padded to 96 (3*32)
#define DCATP   288         // 3 * DINP
#define ROWS    (BATCH*NPTS)
#define ALPHA_C 0.05f
#define BETA_C  0.95f
#define GAMMA_C 0.95f
#define INV_S2  0.70710678118654752440f

// ---------------------------------------------------------------------------
// WMMA fragment helpers (V_WMMA_F32_16X16X32_BF16, wave32)
//   A (16x32): row m=lane&15; kg=lane>>4; e0..7 = K kg*8.., e8..15 = K 16+kg*8..
//   B (32x16): col n=lane&15; e0..7 = K kg*8.., e8..15 = K 16+kg*8.. (needs K-contiguous src)
//   C/D f32 : element r -> row r + 8*kg, col = lane&15
// ---------------------------------------------------------------------------
__device__ __forceinline__ v16bf load_a_frag(const __bf16* __restrict__ base_row,
                                             int kg) {
  v8bf lo = *(const v8bf*)(base_row + kg * 8);
  v8bf hi = *(const v8bf*)(base_row + 16 + kg * 8);
  v16bf a;
#pragma unroll
  for (int i = 0; i < 8; ++i) { a[i] = lo[i]; a[8 + i] = hi[i]; }
  return a;
}

__device__ __forceinline__ v16bf load_b_frag(const __bf16* __restrict__ colp,
                                             int kg) {
  v8bf b0 = *(const v8bf*)(colp + kg * 8);
  v8bf b1 = *(const v8bf*)(colp + 16 + kg * 8);
  v16bf b;
#pragma unroll
  for (int i = 0; i < 8; ++i) { b[i] = b0[i]; b[8 + i] = b1[i]; }
  return b;
}

// ---------------------------------------------------------------------------
// Graph diffusion, batch b:  out = Mt(512x512) @ h_segin(512x96) + alpha*h0
//   catCM: [BATCH][DCATP][NPTS] bf16 (channel-major; B operand, K-contiguous)
//   catRM: [BATCH][NPTS][DCATP] bf16 (row-major; projection A operand)
// grid (NPTS/128, BATCH), block 256 (8 waves)
// ---------------------------------------------------------------------------
__global__ __launch_bounds__(256)
void diffusion_kernel(const __bf16* __restrict__ Mt, __bf16* __restrict__ catCM,
                      __bf16* __restrict__ catRM, int seg_in, int seg_out,
                      int write_cm) {
  const int b    = blockIdx.y;
  const int tid  = threadIdx.x;
  const int wave = tid >> 5;
  const int lane = tid & 31;
  const int m    = lane & 15;
  const int kg   = lane >> 4;
  const int row0 = blockIdx.x * 128 + wave * 16;

  __shared__ __attribute__((aligned(16))) __bf16 ldsB[DINP * 32]; // [c][k], 6 KB
  __bf16* cmb = catCM + (size_t)b * DCATP * NPTS;
  __bf16* rmb = catRM + (size_t)b * NPTS * DCATP;

  v8f acc[6];
#pragma unroll
  for (int j = 0; j < 6; ++j)
#pragma unroll
    for (int e = 0; e < 8; ++e) acc[j][e] = 0.0f;

  for (int kc = 0; kc < NPTS / 32; ++kc) {
    const int k0 = kc * 32;
    __syncthreads();
    // wide staging: 96 channels x 32 nodes, 16B granules both sides
#pragma unroll
    for (int idx = tid; idx < DINP * 4; idx += 256) {   // 384 granules
      int c = idx >> 2, part = (idx & 3) * 8;
      *(v8bf*)(ldsB + c * 32 + part) =
          *(const v8bf*)(cmb + (size_t)(seg_in + c) * NPTS + k0 + part);
    }
    __syncthreads();

    const __bf16* ap = Mt + (size_t)(row0 + m) * NPTS + k0;
    if (kc + 1 < NPTS / 32) __builtin_prefetch(ap + 32, 0, 1);
    v16bf afrag = load_a_frag(ap, kg);

#pragma unroll
    for (int j = 0; j < 6; ++j) {
      v16bf bfrag = load_b_frag(ldsB + (j * 16 + m) * 32, kg);
      acc[j] = __builtin_amdgcn_wmma_f32_16x16x32_bf16(
          false, afrag, false, bfrag, (short)0, acc[j], false, false);
    }
  }

  // epilogue: += alpha*h0 (seg0, CM: 8 consecutive rows at fixed channel)
#pragma unroll
  for (int j = 0; j < 6; ++j) {
    const int gc = j * 16 + m;
    const int gr0 = row0 + 8 * kg;
    v8bf h0 = *(const v8bf*)(cmb + (size_t)gc * NPTS + gr0);
    v8bf pk;
#pragma unroll
    for (int r = 0; r < 8; ++r) {
      float v = acc[j][r] + ALPHA_C * (float)h0[r];
      pk[r] = (__bf16)v;
      rmb[(size_t)(gr0 + r) * DCATP + seg_out + gc] = pk[r]; // coalesced across lanes
    }
    if (write_cm)
      *(v8bf*)(cmb + (size_t)(seg_out + gc) * NPTS + gr0) = pk;
  }
}

// ---------------------------------------------------------------------------
// Projection: out(ROWS x NCOL) = catRM(ROWS x 288) @ W + bias, + activation
//   Wt: transposed weights [NCOL][288] (K-contiguous B operand)
// act 0: sigmoid, cols [0,64)->out0 (z), [64,128)->out1 (r);  act 1: tanh->out0
// grid ROWS/128, block 256 (8 waves)
// ---------------------------------------------------------------------------
template <int NCOL>
__global__ __launch_bounds__(256)
void proj_kernel(const __bf16* __restrict__ catRM, const __bf16* __restrict__ Wt,
                 const float* __restrict__ bias, float* __restrict__ out0,
                 float* __restrict__ out1, int act) {
  const int tid  = threadIdx.x;
  const int wave = tid >> 5;
  const int lane = tid & 31;
  const int m    = lane & 15;
  const int kg   = lane >> 4;
  const int row0 = blockIdx.x * 128 + wave * 16;
  constexpr int NT = NCOL / 16;

  __shared__ __attribute__((aligned(16))) __bf16 ldsW[NCOL * 32]; // [c][k]

  v8f acc[NT];
#pragma unroll
  for (int j = 0; j < NT; ++j)
#pragma unroll
    for (int e = 0; e < 8; ++e) acc[j][e] = 0.0f;

  for (int kc = 0; kc < DCATP / 32; ++kc) {
    const int k0 = kc * 32;
    __syncthreads();
#pragma unroll
    for (int idx = tid; idx < NCOL * 4; idx += 256) {   // 16B granules
      int c = idx >> 2, part = (idx & 3) * 8;
      *(v8bf*)(ldsW + c * 32 + part) =
          *(const v8bf*)(Wt + (size_t)c * DCATP + k0 + part);
    }
    __syncthreads();

    const __bf16* ap = catRM + (size_t)(row0 + m) * DCATP + k0;
    v16bf afrag = load_a_frag(ap, kg);

#pragma unroll
    for (int j = 0; j < NT; ++j) {
      v16bf bfrag = load_b_frag(ldsW + (j * 16 + m) * 32, kg);
      acc[j] = __builtin_amdgcn_wmma_f32_16x16x32_bf16(
          false, afrag, false, bfrag, (short)0, acc[j], false, false);
    }
  }

#pragma unroll
  for (int j = 0; j < NT; ++j) {
#pragma unroll
    for (int r = 0; r < 8; ++r) {
      int gr = row0 + r + 8 * kg;
      int gc = j * 16 + m;
      float v = acc[j][r] + bias[gc];
      if (act == 0) {
        float s = 1.0f / (1.0f + __expf(-v));
        if (gc < 64) out0[(size_t)gr * 64 + gc] = s;
        else         out1[(size_t)gr * 64 + (gc - 64)] = s;
      } else {
        out0[(size_t)gr * 64 + gc] = tanhf(v);
      }
    }
  }
}

// ---------------------------------------------------------------------------
// cat seg0 assembly into BOTH layouts: [xt(4) | H or r*H (64) | pad 0]
// (CM stores are coalesced: consecutive lanes = consecutive nodes)
// mode 0: encoder D (xe-xo), 1: encoder A (xe+xo), 2: decoder
// ---------------------------------------------------------------------------
__global__ void assemble_kernel(const float* __restrict__ x,
                                const float* __restrict__ st,
                                const float* __restrict__ H,
                                const float* __restrict__ rg,
                                const float* __restrict__ decin,
                                __bf16* __restrict__ catCM,
                                __bf16* __restrict__ catRM, int t, int mode) {
  int row = blockIdx.x * blockDim.x + threadIdx.x;
  if (row >= ROWS) return;
  int b = row >> 9, n = row & 511;

  float xt[4];
  if (mode < 2) {
    int tt = 2 * t;
#pragma unroll
    for (int c = 0; c < 3; ++c) {
      float xe = x[(((size_t)b * 3 + c) * NPTS + n) * SEQLEN + tt];
      float xo = x[(((size_t)b * 3 + c) * NPTS + n) * SEQLEN + tt + 1];
      xt[c] = (mode ? (xe + xo) : (xe - xo)) * INV_S2;
    }
    float hour   = (st[((size_t)b * 5 + 3) * SEQLEN + tt] + 0.5f) * 23.0f;
    float minute = (st[((size_t)b * 5 + 4) * SEQLEN + tt] + 0.5f) * 59.0f;
    xt[3] = (float)(int)((hour * 60.0f + minute) / 5.0f);   // 60/PPH = 5
  } else {
#pragma unroll
    for (int c = 0; c < 3; ++c) xt[c] = decin[((size_t)b * 3 + c) * NPTS + n];
    xt[3] = (float)(int)((st[((size_t)b * 5 + 2) * SEQLEN + t] + 0.5f) * 6.0f);
  }

  __bf16* rrow_out = catRM + (size_t)row * DCATP;
  __bf16* cmb      = catCM + (size_t)b * DCATP * NPTS + n;
#pragma unroll
  for (int c = 0; c < 4; ++c) {
    __bf16 v = (__bf16)xt[c];
    rrow_out[c] = v;
    cmb[(size_t)c * NPTS] = v;
  }
  const float* hrow = H + (size_t)row * HIDC;
  if (rg) {
    const float* rr = rg + (size_t)row * HIDC;
    for (int k = 0; k < HIDC; ++k) {
      __bf16 v = (__bf16)(hrow[k] * rr[k]);
      rrow_out[4 + k] = v;
      cmb[(size_t)(4 + k) * NPTS] = v;
    }
  } else {
    for (int k = 0; k < HIDC; ++k) {
      __bf16 v = (__bf16)hrow[k];
      rrow_out[4 + k] = v;
      cmb[(size_t)(4 + k) * NPTS] = v;
    }
  }
  for (int k = 68; k < DINP; ++k) rrow_out[k] = (__bf16)0.0f;
}

// zero the CM seg0 pad channels (68..95) once per launch
__global__ void zeropad_kernel(__bf16* __restrict__ catCM) {
  int i = blockIdx.x * blockDim.x + threadIdx.x;
  const int per_b = (DINP - 68) * NPTS;             // 28*512
  if (i >= BATCH * per_b) return;
  int b = i / per_b, r = i % per_b;
  catCM[(size_t)b * DCATP * NPTS + (size_t)68 * NPTS + r] = (__bf16)0.0f;
}

// H = z*H + (1-z)*C
__global__ void update_kernel(float* __restrict__ H, const float* __restrict__ z,
                              const float* __restrict__ C, int n) {
  int i = blockIdx.x * blockDim.x + threadIdx.x;
  if (i >= n) return;
  float zz = z[i];
  H[i] = zz * H[i] + (1.0f - zz) * C[i];
}

// ---------------------------------------------------------------------------
// prep: row-sum of (A+I);  Mt = (beta*Anorm + gamma*A)^T in bf16
// ---------------------------------------------------------------------------
__global__ void rowsum_kernel(const float* __restrict__ A, float* __restrict__ rinv) {
  int v = blockIdx.x, lane = threadIdx.x;
  float s = 0.0f;
  for (int w = lane; w < NPTS; w += 32)
    s += A[(size_t)v * NPTS + w] + ((v == w) ? 1.0f : 0.0f);
  for (int off = 16; off >= 1; off >>= 1) s += __shfl_xor(s, off, 32);
  if (lane == 0) rinv[v] = 1.0f / s;
}

__global__ void mt_kernel(const float* __restrict__ A, const float* __restrict__ rinv,
                          __bf16* __restrict__ Mt) {
  int i = blockIdx.x * blockDim.x + threadIdx.x;
  if (i >= NPTS * NPTS) return;
  int r = i >> 9, k = i & 511;                 // Mt[r,k] = M[k,r]
  float a = A[(size_t)k * NPTS + r];
  float an = (a + ((k == r) ? 1.0f : 0.0f)) * rinv[k];
  Mt[i] = (__bf16)(BETA_C * an + GAMMA_C * a);
}

// fused+padded+TRANSPOSED gate weights: Wzrt[3][128][288], Wct[3][64][288]
__global__ void wprep_kernel(const float* __restrict__ Wg, __bf16* __restrict__ Wzrt,
                             __bf16* __restrict__ Wct) {
  int i = blockIdx.x * blockDim.x + threadIdx.x;
  const int total = 3 * DCATP * 192;
  if (i >= total) return;
  int base = i / (DCATP * 192);
  int rem  = i % (DCATP * 192);
  int k    = rem / 192;
  int col  = rem % 192;
  int seg = k / DINP, off = k % DINP;
  float v = 0.0f;
  if (off < 68) {
    int r = seg * 68 + off;
    int gate = col >> 6;      // 0:z 1:r 2:c
    int cc = col & 63;
    const float* w0 = Wg + ((size_t)(6 * base + 2 * gate) * 204 + r) * 64 + cc;
    v = w0[0] + w0[204 * 64];
  }
  if (col < 128) Wzrt[((size_t)base * 128 + col) * DCATP + k] = (__bf16)v;
  else           Wct[((size_t)base * 64 + (col - 128)) * DCATP + k] = (__bf16)v;
}

__global__ void bprep_kernel(const float* __restrict__ bg, float* __restrict__ bzr,
                             float* __restrict__ bc) {
  int i = blockIdx.x * blockDim.x + threadIdx.x;
  if (i >= 3 * 192) return;
  int base = i / 192, col = i % 192;
  int gate = col >> 6, cc = col & 63;
  float v = bg[(6 * base + 2 * gate) * 64 + cc] + bg[(6 * base + 2 * gate + 1) * 64 + cc];
  if (col < 128) bzr[base * 128 + col] = v;
  else           bc[base * 64 + cc] = v;
}

// ---------------------------------------------------------------------------
// mixing (idwt + mix fused; _C branch of the reference is dead code)
// ---------------------------------------------------------------------------
__global__ void mix_kernel(const float* __restrict__ H1, const float* __restrict__ H2,
                           const float* __restrict__ Wmix, const float* __restrict__ bmix,
                           const float* __restrict__ Widwt, const float* __restrict__ bidwt,
                           float* __restrict__ Hd) {
  int row = blockIdx.x;
  int c = threadIdx.x;
  __shared__ float s[64], lo[64], hi[64], h1s[64], h2s[64];
  h1s[c] = H1[(size_t)row * 64 + c];
  h2s[c] = H2[(size_t)row * 64 + c];
  s[c] = h1s[c] + h2s[c];
  __syncthreads();
  float l = 2.0f * bidwt[c], h = 2.0f * bidwt[64 + c];
  for (int j = 0; j < 64; ++j) {
    l += s[j] * Widwt[j * 64 + c];
    h += s[j] * Widwt[64 * 64 + j * 64 + c];
  }
  lo[c] = l; hi[c] = h;
  __syncthreads();
  const float* Wm1 = Wmix + 1 * 128 * 64;
  const float* Wm2 = Wmix + 2 * 128 * 64;
  float hm = bmix[128 + c], hc = bmix[64 + c];
  for (int k = 0; k < 64; ++k) {
    hm += lo[k] * Wm2[(2 * k) * 64 + c] + hi[k] * Wm2[(2 * k + 1) * 64 + c];
    hc += h1s[k] * Wm1[k * 64 + c] + h2s[k] * Wm1[(64 + k) * 64 + c];
  }
  Hd[(size_t)row * 64 + c] = 0.3f * hm + 0.7f * hc;
}

// decoder FC (64->3) + reference output index permutation
__global__ void fc_kernel(const float* __restrict__ H, const float* __restrict__ Wfc,
                          const float* __restrict__ bfc, float* __restrict__ decin,
                          float* __restrict__ out, int t) {
  int row = blockIdx.x * blockDim.x + threadIdx.x;
  if (row >= ROWS) return;
  int b = row >> 9, n = row & 511;
  const float* hrow = H + (size_t)row * 64;
#pragma unroll
  for (int o = 0; o < 3; ++o) {
    float v = bfc[o];
    for (int k = 0; k < 64; ++k) v += hrow[k] * Wfc[k * 3 + o];
    decin[((size_t)b * 3 + o) * NPTS + n] = v;
    int kk = t * 3 + o, od = kk / 12, ol = kk % 12;
    out[(((size_t)b * 3 + od) * NPTS + n) * OUTLEN + ol] = v;
  }
}

// ---------------------------------------------------------------------------
static inline size_t align256(size_t x) { return (x + 255) & ~(size_t)255; }

extern "C" void kernel_launch(void* const* d_in, const int* in_sizes, int n_in,
                              void* d_out, int out_size, void* d_ws, size_t ws_size,
                              hipStream_t stream) {
  const float* x     = (const float*)d_in[0];
  const float* st    = (const float*)d_in[2];
  const float* A     = (const float*)d_in[3];
  const float* Wg    = (const float*)d_in[4];
  const float* bg    = (const float*)d_in[5];
  const float* Wmix  = (const float*)d_in[6];
  const float* bmix  = (const float*)d_in[7];
  const float* Widwt = (const float*)d_in[8];
  const float* bidwt = (const float*)d_in[9];
  const float* Wfc   = (const float*)d_in[10];
  const float* bfc   = (const float*)d_in[11];
  float* out = (float*)d_out;

  char* p = (char*)d_ws;
  auto carve = [&](size_t bytes) { char* r = p; p += align256(bytes); return r; };
  __bf16* Mt    = (__bf16*)carve((size_t)NPTS * NPTS * 2);
  float*  rinv  = (float*)carve(NPTS * 4);
  __bf16* Wzrt  = (__bf16*)carve((size_t)3 * 128 * DCATP * 2);
  __bf16* Wct   = (__bf16*)carve((size_t)3 * 64 * DCATP * 2);
  float*  bzr   = (float*)carve(3 * 128 * 4);
  float*  bc    = (float*)carve(3 * 64 * 4);
  __bf16* catCM = (__bf16*)carve((size_t)BATCH * DCATP * NPTS * 2);
  __bf16* catRM = (__bf16*)carve((size_t)ROWS * DCATP * 2);
  float*  zbuf  = (float*)carve((size_t)ROWS * HIDC * 4);
  float*  rbuf  = (float*)carve((size_t)ROWS * HIDC * 4);
  float*  cbuf  = (float*)carve((size_t)ROWS * HIDC * 4);
  float*  H1    = (float*)carve((size_t)ROWS * HIDC * 4);
  float*  H2    = (float*)carve((size_t)ROWS * HIDC * 4);
  float*  Hd    = (float*)carve((size_t)ROWS * HIDC * 4);
  float*  decin = (float*)carve((size_t)BATCH * 3 * NPTS * 4);

  // ---- prep ----
  rowsum_kernel<<<NPTS, 32, 0, stream>>>(A, rinv);
  mt_kernel<<<(NPTS * NPTS + 255) / 256, 256, 0, stream>>>(A, rinv, Mt);
  wprep_kernel<<<(3 * DCATP * 192 + 255) / 256, 256, 0, stream>>>(Wg, Wzrt, Wct);
  bprep_kernel<<<3, 256, 0, stream>>>(bg, bzr, bc);
  zeropad_kernel<<<(BATCH * 28 * NPTS + 255) / 256, 256, 0, stream>>>(catCM);
  hipMemsetAsync(H1, 0, (size_t)ROWS * HIDC * 4, stream);
  hipMemsetAsync(H2, 0, (size_t)ROWS * HIDC * 4, stream);

  const dim3 dgrid(NPTS / 128, BATCH);
  const int  pgrid = ROWS / 128;
  const int  agrid = ROWS / 256;
  const int  ugrid = (ROWS * HIDC + 255) / 256;

  auto gru_step = [&](float* H, int base, int t, int mode) {
    // z,r from comb = [xt, H]
    assemble_kernel<<<agrid, 256, 0, stream>>>(x, st, H, nullptr, decin, catCM, catRM, t, mode);
    diffusion_kernel<<<dgrid, 256, 0, stream>>>(Mt, catCM, catRM, 0, DINP, 1);
    diffusion_kernel<<<dgrid, 256, 0, stream>>>(Mt, catCM, catRM, DINP, 2 * DINP, 0);
    proj_kernel<128><<<pgrid, 256, 0, stream>>>(catRM, Wzrt + (size_t)base * 128 * DCATP,
                                                bzr + base * 128, zbuf, rbuf, 0);
    // C from temp = [xt, r*H]
    assemble_kernel<<<agrid, 256, 0, stream>>>(x, st, H, rbuf, decin, catCM, catRM, t, mode);
    diffusion_kernel<<<dgrid, 256, 0, stream>>>(Mt, catCM, catRM, 0, DINP, 1);
    diffusion_kernel<<<dgrid, 256, 0, stream>>>(Mt, catCM, catRM, DINP, 2 * DINP, 0);
    proj_kernel<64><<<pgrid, 256, 0, stream>>>(catRM, Wct + (size_t)base * 64 * DCATP,
                                               bc + base * 64, cbuf, nullptr, 1);
    update_kernel<<<ugrid, 256, 0, stream>>>(H, zbuf, cbuf, ROWS * HIDC);
  };

  // ---- encoders ----
  for (int t = 0; t < HALF; ++t) gru_step(H1, 0, t, 0);   // detail branch
  for (int t = 0; t < HALF; ++t) gru_step(H2, 1, t, 1);   // approx branch

  // ---- mix ----
  mix_kernel<<<ROWS, 64, 0, stream>>>(H1, H2, Wmix, bmix, Widwt, bidwt, Hd);

  // ---- decoder ----
  hipMemsetAsync(decin, 0, (size_t)BATCH * 3 * NPTS * 4, stream);
  for (int t = 0; t < OUTLEN; ++t) {
    gru_step(Hd, 2, t, 2);
    fc_kernel<<<agrid, 256, 0, stream>>>(Hd, Wfc, bfc, decin, out, t);
  }
}
